// GlmImageVQVAEVectorQuantizer_34694745817245
// MI455X (gfx1250) — matrix operand order repaired and checked by reading, not compiled, not run on hardware
//
#include <hip/hip_runtime.h>
#include <math.h>

// Problem sizes (fixed by the reference)
#define NPIX   16384   // 16 * 32 * 32 pixel vectors
#define KCODES 16384   // codebook entries
#define DDIM   32      // embedding dim
#define HW     1024    // 32*32
#define QELEMS 524288  // 16*32*32*32 quant elements
#define KTILES (KCODES / 16)

typedef __attribute__((ext_vector_type(16))) __bf16         v16bf;
typedef __attribute__((ext_vector_type(8)))  float          v8f;
typedef __attribute__((ext_vector_type(16))) unsigned short v16us;
typedef __attribute__((ext_vector_type(8)))  unsigned short v8us;

__device__ __forceinline__ unsigned short f2bf(float x) {
    unsigned u = __float_as_uint(x);
    unsigned r = u + 0x7fffu + ((u >> 16) & 1u);   // round-to-nearest-even
    return (unsigned short)(r >> 16);
}
__device__ __forceinline__ float bf2f(unsigned short h) {
    return __uint_as_float(((unsigned)h) << 16);
}

// ---------------------------------------------------------------------------
// K1: normalize hidden_state (NCHW gather), emit fp32 + bf16 hi/lo planes.
// ---------------------------------------------------------------------------
__global__ __launch_bounds__(256)
void vq_norm_hidden(const float* __restrict__ hid,
                    float* __restrict__ Xn,
                    unsigned short* __restrict__ Xh,
                    unsigned short* __restrict__ Xl,
                    float* __restrict__ lossAcc) {
    int n = blockIdx.x * blockDim.x + threadIdx.x;      // pixel id
    if (n == 0) lossAcc[0] = 0.0f;                      // per-launch reset
    int b  = n >> 10;
    int hw = n & (HW - 1);
    const float* base = hid + (size_t)b * (DDIM * HW) + hw;

    float v[DDIM];
    float s = 0.0f;
#pragma unroll
    for (int c = 0; c < DDIM; ++c) {                    // coalesced over hw
        float x = base[c * HW];
        v[c] = x;
        s += x * x;
    }
    float scale = 1.0f / fmaxf(sqrtf(s), 1e-12f);

    float*          xn = Xn + (size_t)n * DDIM;
    unsigned short* xh = Xh + (size_t)n * DDIM;
    unsigned short* xl = Xl + (size_t)n * DDIM;
#pragma unroll
    for (int c = 0; c < DDIM; ++c) {
        float x = v[c] * scale;
        xn[c] = x;
        unsigned short h = f2bf(x);
        xh[c] = h;
        xl[c] = f2bf(x - bf2f(h));                      // residual -> lo plane
    }
}

// ---------------------------------------------------------------------------
// K2: normalize codebook rows, emit fp32 + bf16 hi/lo planes.
// ---------------------------------------------------------------------------
__global__ __launch_bounds__(256)
void vq_norm_embed(const float* __restrict__ emb,
                   float* __restrict__ Enf,
                   unsigned short* __restrict__ Eh,
                   unsigned short* __restrict__ El) {
    int k = blockIdx.x * blockDim.x + threadIdx.x;      // codebook row
    const float4* row = (const float4*)(emb + (size_t)k * DDIM);
    float v[DDIM];
    float s = 0.0f;
#pragma unroll
    for (int q = 0; q < DDIM / 4; ++q) {
        float4 f = row[q];
        v[4*q+0] = f.x; v[4*q+1] = f.y; v[4*q+2] = f.z; v[4*q+3] = f.w;
        s += f.x*f.x + f.y*f.y + f.z*f.z + f.w*f.w;
    }
    float scale = 1.0f / fmaxf(sqrtf(s), 1e-12f);

    float*          ef = Enf + (size_t)k * DDIM;
    unsigned short* eh = Eh  + (size_t)k * DDIM;
    unsigned short* el = El  + (size_t)k * DDIM;
#pragma unroll
    for (int c = 0; c < DDIM; ++c) {
        float x = v[c] * scale;
        ef[c] = x;
        unsigned short h = f2bf(x);
        eh[c] = h;
        el[c] = f2bf(x - bf2f(h));
    }
}

// ---------------------------------------------------------------------------
// K3: fused GEMM + argmin via V_WMMA_F32_16X16X32_BF16 (hi/lo compensated).
// One wave owns 16 X-rows and streams all 1024 codebook tiles.
// Unroll-by-2 ping-pong: each buffer's refill loads are issued right AFTER
// the WMMAs that consume it (WAR pins the order), so they overlap the other
// buffer's 3 WMMAs + argmin compares -- latency hiding the scheduler cannot
// collapse. Eh/El carry 2 padding tiles so lookahead loads (up to kt+3) are
// branch-free; padded data is loaded but never consumed.
// ---------------------------------------------------------------------------
__global__ __launch_bounds__(256)
void vq_argmin_wmma(const unsigned short* __restrict__ Xh,
                    const unsigned short* __restrict__ Xl,
                    const unsigned short* __restrict__ Eh,
                    const unsigned short* __restrict__ El,
                    int* __restrict__ bestIdxOut,
                    float* __restrict__ idxOutF) {
    const int lid     = threadIdx.x & 31;
    const int wave    = blockIdx.x * (blockDim.x >> 5) + (threadIdx.x >> 5);
    const int rowBase = wave << 4;                      // 16 X-rows per wave
    const int sub     = lid & 15;
    const int half    = lid >> 4;

    // --- A tile (16x32 bf16): lane layout: M = sub, K-base = half*16,
    //     elements [kb..kb+8) and [kb+8..kb+16) -> two aligned 16B runs.
    const int M   = rowBase + sub;
    const int kbA = half * 16;
    const v8us* ah0 = (const v8us*)(Xh + (size_t)M * DDIM + kbA);
    const v8us* ah1 = (const v8us*)(Xh + (size_t)M * DDIM + kbA + 8);
    const v8us* al0 = (const v8us*)(Xl + (size_t)M * DDIM + kbA);
    const v8us* al1 = (const v8us*)(Xl + (size_t)M * DDIM + kbA + 8);
    v8us h0 = *ah0, h1 = *ah1, l0 = *al0, l1 = *al1;
    v16us Ah, Al;
#pragma unroll
    for (int i = 0; i < 8; ++i) {
        Ah[i] = h0[i]; Ah[8 + i] = h1[i];
        Al[i] = l0[i]; Al[8 + i] = l1[i];
    }
    const v16bf Ahb = __builtin_bit_cast(v16bf, Ah);
    const v16bf Alb = __builtin_bit_cast(v16bf, Al);

    // --- B tile base (32x16 bf16): lane = column (codebook row sub within
    //     tile), half-wave selects K half: 16 contiguous bf16 = one 32B load.
    const int col = sub;
    const unsigned short* bhBase = Eh + (size_t)col * DDIM + half * 16;
    const unsigned short* blBase = El + (size_t)col * DDIM + half * 16;

    float best[8];
    int   bidx[8];
#pragma unroll
    for (int j = 0; j < 8; ++j) { best[j] = -INFINITY; bidx[j] = 0; }

    // prologue: buffer A <- tile 0, buffer B <- tile 1
    v16us bhA = *(const v16us*)(bhBase);
    v16us blA = *(const v16us*)(blBase);
    v16us bhB = *(const v16us*)(bhBase + (size_t)16 * DDIM);
    v16us blB = *(const v16us*)(blBase + (size_t)16 * DDIM);

    for (int kt = 0; kt < KTILES; kt += 2) {
        // ---- tile kt (buffer A) ----
        {
            v8f c = {};
            c = __builtin_amdgcn_wmma_f32_16x16x32_bf16(
                    false, Ahb, false, __builtin_bit_cast(v16bf, bhA),
                    (short)0, c, false, false);
            c = __builtin_amdgcn_wmma_f32_16x16x32_bf16(
                    false, Ahb, false, __builtin_bit_cast(v16bf, blA),
                    (short)0, c, false, false);
            c = __builtin_amdgcn_wmma_f32_16x16x32_bf16(
                    false, Alb, false, __builtin_bit_cast(v16bf, bhA),
                    (short)0, c, false, false);
            const int kidx = kt * 16 + col;
#pragma unroll
            for (int j = 0; j < 8; ++j)
                if (c[j] > best[j]) { best[j] = c[j]; bidx[j] = kidx; }
        }
        // refill A with tile kt+2 (WAR on bhA/blA keeps these after the
        // WMMAs above; they overlap buffer B's WMMAs + compares below)
        bhA = *(const v16us*)(bhBase + (size_t)(kt + 2) * 16 * DDIM);
        blA = *(const v16us*)(blBase + (size_t)(kt + 2) * 16 * DDIM);

        // ---- tile kt+1 (buffer B) ----
        {
            v8f c = {};
            c = __builtin_amdgcn_wmma_f32_16x16x32_bf16(
                    false, Ahb, false, __builtin_bit_cast(v16bf, bhB),
                    (short)0, c, false, false);
            c = __builtin_amdgcn_wmma_f32_16x16x32_bf16(
                    false, Ahb, false, __builtin_bit_cast(v16bf, blB),
                    (short)0, c, false, false);
            c = __builtin_amdgcn_wmma_f32_16x16x32_bf16(
                    false, Alb, false, __builtin_bit_cast(v16bf, bhB),
                    (short)0, c, false, false);
            const int kidx = (kt + 1) * 16 + col;
#pragma unroll
            for (int j = 0; j < 8; ++j)
                if (c[j] > best[j]) { best[j] = c[j]; bidx[j] = kidx; }
        }
        // refill B with tile kt+3
        bhB = *(const v16us*)(bhBase + (size_t)(kt + 3) * 16 * DDIM);
        blB = *(const v16us*)(blBase + (size_t)(kt + 3) * 16 * DDIM);
    }

    // --- argmax reduction within each 16-lane half (ties -> smaller index)
#pragma unroll
    for (int m = 1; m <= 8; m <<= 1) {
#pragma unroll
        for (int j = 0; j < 8; ++j) {
            float ov = __shfl_xor(best[j], m, 32);
            int   oi = __shfl_xor(bidx[j], m, 32);
            if (ov > best[j] || (ov == best[j] && oi < bidx[j])) {
                best[j] = ov; bidx[j] = oi;
            }
        }
    }

    if (sub < 8) {                                      // after all WMMAs
        int row = rowBase + half * 8 + sub;
        int bi  = bidx[sub];
        bestIdxOut[row] = bi;
        idxOutF[row]    = (float)bi;
    }
}

// ---------------------------------------------------------------------------
// K4: gather quant (NHWC->NCHW scatter) + loss partial reduction.
// ---------------------------------------------------------------------------
__global__ __launch_bounds__(256)
void vq_quant_loss(const int* __restrict__ bestIdx,
                   const float* __restrict__ Enf,
                   const float* __restrict__ Xn,
                   float* __restrict__ outQuant,
                   float* __restrict__ lossAcc) {
    int n  = blockIdx.x * blockDim.x + threadIdx.x;
    int b  = n >> 10;
    int hw = n & (HW - 1);
    int idx = bestIdx[n];
    const float4* q4 = (const float4*)(Enf + (size_t)idx * DDIM);
    const float4* x4 = (const float4*)(Xn + (size_t)n * DDIM);
    float* outBase = outQuant + (size_t)b * (DDIM * HW) + hw;

    float local = 0.0f;
#pragma unroll
    for (int q = 0; q < DDIM / 4; ++q) {
        float4 e = q4[q];
        float4 x = x4[q];
        float d0 = e.x - x.x, d1 = e.y - x.y, d2 = e.z - x.z, d3 = e.w - x.w;
        local += d0*d0 + d1*d1 + d2*d2 + d3*d3;
        outBase[(4*q + 0) * HW] = e.x;                  // coalesced over hw
        outBase[(4*q + 1) * HW] = e.y;
        outBase[(4*q + 2) * HW] = e.z;
        outBase[(4*q + 3) * HW] = e.w;
    }
#pragma unroll
    for (int m = 16; m >= 1; m >>= 1) local += __shfl_xor(local, m, 32);
    if ((threadIdx.x & 31) == 0) atomicAdd(lossAcc, local);
}

// ---------------------------------------------------------------------------
// K5: finalize loss = (1 + BETA) * mean
// ---------------------------------------------------------------------------
__global__ void vq_finalize(const float* __restrict__ lossAcc,
                            float* __restrict__ lossOut) {
    lossOut[0] = 1.25f * lossAcc[0] * (1.0f / (float)QELEMS);
}

// ---------------------------------------------------------------------------
extern "C" void kernel_launch(void* const* d_in, const int* in_sizes, int n_in,
                              void* d_out, int out_size, void* d_ws, size_t ws_size,
                              hipStream_t stream) {
    const float* hid = (const float*)d_in[0];   // [16,32,32,32] fp32
    const float* emb = (const float*)d_in[1];   // [16384,32]    fp32
    float* out = (float*)d_out;                 // quant | loss | indices

    // workspace carve (all power-of-two aligned); Eh/El padded by two 16-row
    // tiles (32 rows) for the branch-free ping-pong lookahead in K3.
    char* w = (char*)d_ws;
    float*          Xn  = (float*)w;          w += (size_t)NPIX   * DDIM * 4;
    float*          Enf = (float*)w;          w += (size_t)KCODES * DDIM * 4;
    unsigned short* Xh  = (unsigned short*)w; w += (size_t)NPIX   * DDIM * 2;
    unsigned short* Xl  = (unsigned short*)w; w += (size_t)NPIX   * DDIM * 2;
    unsigned short* Eh  = (unsigned short*)w; w += ((size_t)KCODES + 32) * DDIM * 2;
    unsigned short* El  = (unsigned short*)w; w += ((size_t)KCODES + 32) * DDIM * 2;
    int*            bestIdx = (int*)w;        w += (size_t)NPIX * 4;
    float*          lossAcc = (float*)w;

    float* outQuant = out;                 // [524288]
    float* outLoss  = out + QELEMS;        // [1]
    float* outIdx   = out + QELEMS + 1;    // [16384]

    vq_norm_hidden<<<NPIX / 256, 256, 0, stream>>>(hid, Xn, Xh, Xl, lossAcc);
    vq_norm_embed <<<KCODES / 256, 256, 0, stream>>>(emb, Enf, Eh, El);
    // 1024 waves total: 128 blocks x 8 waves; each wave owns a 16-row tile
    vq_argmin_wmma<<<(NPIX / 16) / 8, 256, 0, stream>>>(Xh, Xl, Eh, El,
                                                        bestIdx, outIdx);
    vq_quant_loss <<<NPIX / 256, 256, 0, stream>>>(bestIdx, Enf, Xn,
                                                   outQuant, lossAcc);
    vq_finalize   <<<1, 1, 0, stream>>>(lossAcc, outLoss);
}
